// FFTMemAutoEncoderBranch_39204461478045
// MI455X (gfx1250) — matrix operand order repaired and checked by reading, not compiled, not run on hardware
//
#include <hip/hip_runtime.h>
#include <hip/hip_bf16.h>
#include <math.h>

// ---------------------------------------------------------------------------
// FFT-magnitude memory autoencoder branch for MI455X (gfx1250, wave32).
// Conv2/Conv3/sim: implicit GEMM on v_wmma_f32_16x16x32_f16.
// Conv2 interior A-tiles: Tensor Data Mover (tensor_load_to_lds) when the
// builtin is available; vectorized global->LDS fallback otherwise.
// ---------------------------------------------------------------------------

typedef __attribute__((ext_vector_type(16))) _Float16 v16h;
typedef __attribute__((ext_vector_type(8)))  _Float16 v8h;
typedef __attribute__((ext_vector_type(8)))  float    v8f;
typedef __attribute__((ext_vector_type(4)))  unsigned int u32x4;
typedef __attribute__((ext_vector_type(8)))  int      i32x8;
typedef __attribute__((ext_vector_type(4)))  int      i32x4;

// ---- TDM availability / arity dispatch (see CDNA5_HIP.md) ------------------
#if defined(__has_builtin)
#  if __has_builtin(__builtin_amdgcn_tensor_load_to_lds) && __has_builtin(__builtin_amdgcn_s_wait_tensorcnt)
#    define HAVE_TDM 1
#  endif
#endif
#ifndef HAVE_TDM
#  define HAVE_TDM 0
#endif
#if HAVE_TDM
#  if __has_include(<hip/amd_detail/amd_gfx1250_TDM.h>)
#    define TDM_SIX_ARGS 1
#  else
#    define TDM_SIX_ARGS 0
#  endif
__device__ __forceinline__ void tdm_load_2d_and_wait(u32x4 g0, i32x8 g1) {
  i32x4 z4 = {};
#  if TDM_SIX_ARGS
  i32x8 z8 = {};
  __builtin_amdgcn_tensor_load_to_lds(g0, g1, z4, z4, z8, 0);
#  else
  __builtin_amdgcn_tensor_load_to_lds(g0, g1, z4, z4, 0);
#  endif
  __builtin_amdgcn_s_wait_tensorcnt((short)0);
}
#endif

// ---- WMMA fragment gathers (contiguous k = 16*(lane>>4) + i in A and B, wave32) ----------
// A (f16, 16x32, M x K), tile rows of `stride` f16:
//   lanes 0-15  : M = lane,    K = 0..15
//   lanes 16-31 : M = lane-16, K = 16..31
__device__ __forceinline__ v16h frag_a_from_tile_s(const _Float16* tile, int stride) {
  const int lane = threadIdx.x & 31;
  const int m    = lane & 15;
  const int kb   = (lane >> 4) << 4;            // 0 or 16
  const _Float16* row = tile + m * stride;
  v8h lo = *(const v8h*)(row + kb);             // K = kb .. kb+7
  v8h hi = *(const v8h*)(row + kb + 8);         // K = kb+8 .. kb+15
  v16h r;
#pragma unroll
  for (int i = 0; i < 8; ++i) { r[i] = lo[i]; r[i + 8] = hi[i]; }
  return r;
}

// B (f16, 32x16, K x N) stored TRANSPOSED: tileT[n][k], rows of 32 f16:
//   lanes 0-15: N = lane, K = 0..15; lanes 16-31: N = lane-16, K = 16..31.
// Works on LDS or global (packed weights) pointers.
__device__ __forceinline__ v16h frag_b_from_tileT(const _Float16* tileT) {
  const int lane = threadIdx.x & 31;
  const int n    = lane & 15;
  const int kb   = (lane >> 4) << 4;            // 0 or 16
  return *(const v16h*)(tileT + n * 32 + kb);   // 32-byte contiguous load
}

// ---------------------------------------------------------------------------
// Pass 1: 256-point row FFT (radix-2 DIT in LDS), complex out.
// ---------------------------------------------------------------------------
#define FFT_N 256

__global__ void fft_rows_kernel(const float* __restrict__ x, float2* __restrict__ F) {
  __shared__ float re[FFT_N], im[FFT_N];
  const int row = blockIdx.x;
  const int b   = blockIdx.y;
  const int t   = threadIdx.x;                  // 128 threads
  const float* src = x + ((size_t)b * FFT_N + row) * FFT_N;
  for (int i = t; i < FFT_N; i += 128) {
    int r = (int)(__brev((unsigned)i) >> 24);   // 8-bit bit reversal
    re[r] = src[i];
    im[r] = 0.0f;
  }
  __syncthreads();
#pragma unroll
  for (int s = 1; s <= 8; ++s) {
    const int m  = 1 << s;
    const int mh = m >> 1;
    const int j  = t & (mh - 1);
    const int g  = t >> (s - 1);
    const int i1 = g * m + j;
    const int i2 = i1 + mh;
    float ang = -6.28318530717958647692f * (float)j / (float)m;
    float sn, cs;
    __sincosf(ang, &sn, &cs);
    float xr = re[i2], xi = im[i2];
    float tr = cs * xr - sn * xi;
    float ti = cs * xi + sn * xr;
    float ur = re[i1], ui = im[i1];
    re[i1] = ur + tr; im[i1] = ui + ti;
    re[i2] = ur - tr; im[i2] = ui - ti;
    __syncthreads();
  }
  float2* dst = F + ((size_t)b * FFT_N + row) * FFT_N;
  for (int i = t; i < FFT_N; i += 128) dst[i] = make_float2(re[i], im[i]);
}

// ---------------------------------------------------------------------------
// Pass 2: column FFT + |.| with ortho norm (1/256) + fftshift on both dims.
// ---------------------------------------------------------------------------
__global__ void fft_cols_mag_kernel(const float2* __restrict__ F, float* __restrict__ xm) {
  __shared__ float re[FFT_N], im[FFT_N];
  const int col = blockIdx.x;
  const int b   = blockIdx.y;
  const int t   = threadIdx.x;                  // 128 threads
  const float2* src = F + (size_t)b * FFT_N * FFT_N + col;
  for (int i = t; i < FFT_N; i += 128) {
    int r = (int)(__brev((unsigned)i) >> 24);
    float2 v = src[(size_t)i * FFT_N];
    re[r] = v.x; im[r] = v.y;
  }
  __syncthreads();
#pragma unroll
  for (int s = 1; s <= 8; ++s) {
    const int m  = 1 << s;
    const int mh = m >> 1;
    const int j  = t & (mh - 1);
    const int g  = t >> (s - 1);
    const int i1 = g * m + j;
    const int i2 = i1 + mh;
    float ang = -6.28318530717958647692f * (float)j / (float)m;
    float sn, cs;
    __sincosf(ang, &sn, &cs);
    float xr = re[i2], xi = im[i2];
    float tr = cs * xr - sn * xi;
    float ti = cs * xi + sn * xr;
    float ur = re[i1], ui = im[i1];
    re[i1] = ur + tr; im[i1] = ui + ti;
    re[i2] = ur - tr; im[i2] = ui - ti;
    __syncthreads();
  }
  const int cshift = (col + 128) & 255;
  float* dst = xm + (size_t)b * FFT_N * FFT_N;
  for (int i = t; i < FFT_N; i += 128) {
    int rshift = (i + 128) & 255;
    float mag = sqrtf(re[i] * re[i] + im[i] * im[i]) * (1.0f / 256.0f);
    dst[(size_t)rshift * FFT_N + cshift] = mag;
  }
}

// ---------------------------------------------------------------------------
// Weight packing: f32 [oc][ic][3][3] -> f16 tap-major B^T layout.
// wt2: [9][64][32]   (tap, oc, ic)     wt3: [9][2][128][32] (tap, ch, oc, ic)
// ---------------------------------------------------------------------------
__global__ void pack_w2_kernel(const float* __restrict__ w2, _Float16* __restrict__ wt2) {
  int idx = blockIdx.x * 256 + threadIdx.x;
  if (idx < 9 * 64 * 32) {
    int tap = idx / (64 * 32);
    int r   = idx - tap * (64 * 32);
    int oc = r >> 5, ic = r & 31;
    wt2[idx] = (_Float16)w2[(oc * 32 + ic) * 9 + tap];
  }
}

__global__ void pack_w3_kernel(const float* __restrict__ w3, _Float16* __restrict__ wt3) {
  int idx = blockIdx.x * 256 + threadIdx.x;
  if (idx < 9 * 2 * 128 * 32) {
    int tap = idx / (2 * 128 * 32);
    int r   = idx - tap * (2 * 128 * 32);
    int ch  = r >> 12;                          // 128*32 = 4096
    int r2  = r & 4095;
    int oc = r2 >> 5, ic = r2 & 31;
    wt3[idx] = (_Float16)w3[(oc * 64 + ch * 32 + ic) * 9 + tap];
  }
}

// ---------------------------------------------------------------------------
// Conv1 (1->32, 3x3, pad1) + ReLU + maxpool2 (direct; K=9 too small for WMMA,
// bandwidth-bound).  Output h1 NHWC f16 [B][128][128][32].
// ---------------------------------------------------------------------------
__global__ void conv1_pool_kernel(const float* __restrict__ xm,
                                  const float* __restrict__ w,
                                  const float* __restrict__ bias,
                                  _Float16* __restrict__ h1) {
  __shared__ float wsm[32 * 9];
  __shared__ float bsm[32];
  const int tid = threadIdx.x;                  // 256
  if (tid < 288) wsm[tid] = w[tid];
  if (tid < 32)  bsm[tid] = bias[tid];
  __syncthreads();
  const int b  = blockIdx.z;
  const int px = blockIdx.x * 16 + (tid & 15);  // pooled x: 0..127
  const int py = blockIdx.y * 16 + (tid >> 4);  // pooled y: 0..127
  float p[4][4];
  const float* src = xm + (size_t)b * 256 * 256;
#pragma unroll
  for (int r = 0; r < 4; ++r) {
    int y = 2 * py - 1 + r;
#pragma unroll
    for (int c = 0; c < 4; ++c) {
      int xx = 2 * px - 1 + c;
      p[r][c] = ((unsigned)y < 256u && (unsigned)xx < 256u) ? src[(size_t)y * 256 + xx] : 0.0f;
    }
  }
  _Float16* dst = h1 + (((size_t)b * 128 + py) * 128 + px) * 32;
#pragma unroll 4
  for (int oc = 0; oc < 32; ++oc) {
    float mx = 0.0f;                            // relu => max starts at 0
#pragma unroll
    for (int cy = 0; cy < 2; ++cy)
#pragma unroll
      for (int cx = 0; cx < 2; ++cx) {
        float s = bsm[oc];
#pragma unroll
        for (int dy = 0; dy < 3; ++dy)
#pragma unroll
          for (int dx = 0; dx < 3; ++dx)
            s = fmaf(p[cy + dy][cx + dx], wsm[oc * 9 + dy * 3 + dx], s);
        mx = fmaxf(mx, s);
      }
    dst[oc] = (_Float16)mx;
  }
}

// ---------------------------------------------------------------------------
// Conv2 (32->64) WMMA implicit GEMM + ReLU + fused maxpool2.
// Block: 8x8 conv patch x 64 oc (4 M-tiles x 4 N-tiles); K = 9 taps x 32 ic.
// A-tile per tap = 8 rows x 512 B (stride 8 KB): TDM for interior blocks,
// vectorized v8h copy elsewhere.  B fragment read directly from packed wt2.
// ---------------------------------------------------------------------------
__global__ void conv2_pool_kernel(const _Float16* __restrict__ h1,
                                  const _Float16* __restrict__ wt2,
                                  const float* __restrict__ b2,
                                  _Float16* __restrict__ h2p) {
  __shared__ __align__(32) _Float16 Ah[64][32];   // spatial s x ic
  __shared__ float Cs[64][64];                    // spatial s x oc
  const int tid  = threadIdx.x;                   // 256 = 8 waves
  const int lane = tid & 31;
  const int wv   = tid >> 5;                      // 0..7
  const int b  = blockIdx.z;
  const int x0 = blockIdx.x * 8;
  const int y0 = blockIdx.y * 8;

  const int mt0 = wv >> 2, nt0 = wv & 3;          // combo wv
  const int mt1 = mt0 + 2;                        // combo wv+8 (same N-tile)
  v8f acc0 = {}; v8f acc1 = {};

  // fallback staging assignment: pixel s = tid>>2, 16-byte chunk c = tid&3
  const int s  = tid >> 2;
  const int c  = tid & 3;
  const int sy = s >> 3, sx = s & 7;

  const bool interior = (blockIdx.x >= 1) && (blockIdx.x <= 14) &&
                        (blockIdx.y >= 1) && (blockIdx.y <= 14);
  (void)interior;

  for (int tap = 0; tap < 9; ++tap) {
    const int dy = tap / 3, dx = tap % 3;
    bool used_tdm = false;
#if HAVE_TDM
    if (interior) {
      used_tdm = true;
      if (wv == 0) {
        // D# for a 2D tile: 8 rows x 512 B (64 x 8-byte units), row stride
        // 128 pixels * 64 B = 8192 B (1024 units).  Tile == tensor (in-bounds).
        const size_t ga = (size_t)(const void*)
            (h1 + (((size_t)b * 128 + (y0 + dy - 1)) * 128 + (x0 + dx - 1)) * 32);
        const unsigned lds_addr = (unsigned)(size_t)(void*)&Ah[0][0];
        u32x4 g0;
        g0[0] = 1u;                                   // count=1, user mode
        g0[1] = lds_addr;                             // lds_addr [63:32]
        g0[2] = (unsigned)(ga & 0xFFFFFFFFu);         // global_addr lo
        g0[3] = (unsigned)((ga >> 32) & 0x01FFFFFFu)  // global_addr hi (57b)
              | (2u << 30);                           // type = 2 ("image")
        i32x8 g1;
        g1[0] = (3 << 16);          // data_size = 3 (8 bytes), no flags
        g1[1] = (int)(64u << 16);   // tensor_dim0 = 64 units (low 16 @ bit48)
        g1[2] = (int)(8u << 16);    // tensor_dim0 hi = 0 | tensor_dim1 = 8
        g1[3] = (int)(64u << 16);   // tensor_dim1 hi = 0 | tile_dim0 = 64
        g1[4] = 8;                  // tile_dim1 = 8, tile_dim2 = 0
        g1[5] = 1024;               // tensor_dim0_stride low32 (8192 B / 8)
        g1[6] = 0;                  // stride hi | tensor_dim1_stride lo
        g1[7] = 0;
        tdm_load_2d_and_wait(g0, g1);                 // + s_wait_tensorcnt 0
      }
    }
#endif
    if (!used_tdm) {
      int y  = y0 + sy + dy - 1;
      int xx = x0 + sx + dx - 1;
      v8h v = {};
      if ((unsigned)y < 128u && (unsigned)xx < 128u)
        v = *(const v8h*)(h1 + (((size_t)b * 128 + y) * 128 + xx) * 32 + c * 8);
      *(v8h*)(&Ah[s][c * 8]) = v;
    }
    if (tap < 8)  // cache hint for next tap's activations (global_prefetch_b8)
      __builtin_prefetch(h1 + (((size_t)b * 128 + y0 + sy + tap / 3) * 128 + x0 + sx) * 32, 0, 1);
    __syncthreads();
    const _Float16* wb = wt2 + (size_t)tap * 64 * 32;      // [64][32] B^T slab
    v16h bf = frag_b_from_tileT(wb + nt0 * 16 * 32);       // from global (L2)
    v16h a0 = frag_a_from_tile_s(&Ah[mt0 * 16][0], 32);
    v16h a1 = frag_a_from_tile_s(&Ah[mt1 * 16][0], 32);
    acc0 = __builtin_amdgcn_wmma_f32_16x16x32_f16(false, a0, false, bf, (short)0, acc0, false, false);
    acc1 = __builtin_amdgcn_wmma_f32_16x16x32_f16(false, a1, false, bf, (short)0, acc1, false, false);
    __syncthreads();
  }

  // Bias + ReLU, C tiles -> LDS (f32 C layout: N = lane&15, M = r + 8*(lane>>4))
  {
    const int n  = lane & 15;
    const int oc = nt0 * 16 + n;
    const float bb = b2[oc];
    const int mb = (lane >> 4) * 8;
#pragma unroll
    for (int r = 0; r < 8; ++r) {
      Cs[mt0 * 16 + mb + r][oc] = fmaxf(acc0[r] + bb, 0.0f);
      Cs[mt1 * 16 + mb + r][oc] = fmaxf(acc1[r] + bb, 0.0f);
    }
  }
  __syncthreads();
  // 2x2 maxpool -> 4x4 pooled patch x 64 oc
  for (int idx = tid; idx < 4 * 4 * 64; idx += 256) {
    int oc = idx & 63;
    int p  = idx >> 6;
    int py = p >> 2, px = p & 3;
    int s00 = (2 * py) * 8 + 2 * px;
    float v = fmaxf(fmaxf(Cs[s00][oc], Cs[s00 + 1][oc]),
                    fmaxf(Cs[s00 + 8][oc], Cs[s00 + 9][oc]));
    int gy = blockIdx.y * 4 + py;
    int gx = blockIdx.x * 4 + px;
    h2p[(((size_t)b * 64 + gy) * 64 + gx) * 64 + oc] = (_Float16)v;
  }
}

// ---------------------------------------------------------------------------
// Conv3 (64->128) WMMA + ReLU + fused spatial-sum into q (feat never stored).
// Block: 16 x-positions of one row y, 128 oc (wave w = N-tile w).
// A staged once per tap (16 pixels x 64 ch = one contiguous 2 KB span).
// ---------------------------------------------------------------------------
__global__ void conv3_qsum_kernel(const _Float16* __restrict__ h2p,
                                  const _Float16* __restrict__ wt3,
                                  const float* __restrict__ b3,
                                  float* __restrict__ qsum) {
  __shared__ __align__(32) _Float16 Ah[16][64];
  const int tid  = threadIdx.x;                   // 256 = 8 waves
  const int lane = tid & 31;
  const int wv   = tid >> 5;                      // N-tile 0..7
  const int b  = blockIdx.z;
  const int y  = blockIdx.y;                      // 0..63
  const int x0 = blockIdx.x * 16;                 // 0..48

  const int m = tid >> 3;                         // staging row (tid < 128)
  const int c = tid & 7;                          // 16-byte chunk

  v8f acc = {};
  for (int tap = 0; tap < 9; ++tap) {
    const int dy = tap / 3, dx = tap % 3;
    const int yy = y + dy - 1;
    if (tid < 128) {                              // 16 rows x 8 chunks
      int xx = x0 + m + dx - 1;
      v8h v = {};
      if ((unsigned)yy < 64u && (unsigned)xx < 64u)
        v = *(const v8h*)(h2p + (((size_t)b * 64 + yy) * 64 + xx) * 64 + c * 8);
      *(v8h*)(&Ah[m][c * 8]) = v;
    }
    __syncthreads();
    const _Float16* wb = wt3 + (size_t)tap * 2 * 128 * 32;
#pragma unroll
    for (int ch = 0; ch < 2; ++ch) {
      v16h a  = frag_a_from_tile_s(&Ah[0][ch * 32], 64);
      v16h bf = frag_b_from_tileT(wb + ((size_t)ch * 128 + wv * 16) * 32);
      acc = __builtin_amdgcn_wmma_f32_16x16x32_f16(false, a, false, bf, (short)0, acc, false, false);
    }
    __syncthreads();
  }
  // ReLU + sum over the 16 spatial rows of this tile -> atomic into qsum
  const int n  = lane & 15;
  const int oc = wv * 16 + n;
  const float bb = b3[oc];
  float part = 0.0f;
#pragma unroll
  for (int r = 0; r < 8; ++r) part += fmaxf(acc[r] + bb, 0.0f);
  part += __shfl_down(part, 16);                  // combine M halves (wave32)
  if (lane < 16) atomicAdd(&qsum[(size_t)b * 128 + oc], part);
}

__global__ void zero_kernel(float* __restrict__ p, int n) {
  int i = blockIdx.x * 256 + threadIdx.x;
  if (i < n) p[i] = 0.0f;
}

// q = qsum / 4096, row-normalize -> f16
__global__ void qnorm_kernel(const float* __restrict__ qsum, _Float16* __restrict__ qn) {
  __shared__ float red[128];
  const int b = blockIdx.x, t = threadIdx.x;      // 128 threads
  float v = qsum[(size_t)b * 128 + t] * (1.0f / 4096.0f);
  red[t] = v * v;
  __syncthreads();
  for (int s = 64; s > 0; s >>= 1) { if (t < s) red[t] += red[t + s]; __syncthreads(); }
  float nrm = fmaxf(sqrtf(red[0]), 1e-12f);
  qn[(size_t)b * 128 + t] = (_Float16)(v / nrm);
}

__global__ void knorm_kernel(const float* __restrict__ keys, _Float16* __restrict__ kn) {
  __shared__ float red[128];
  const int m = blockIdx.x, t = threadIdx.x;      // 400 blocks x 128
  float v = keys[(size_t)m * 128 + t];
  red[t] = v * v;
  __syncthreads();
  for (int s = 64; s > 0; s >>= 1) { if (t < s) red[t] += red[t + s]; __syncthreads(); }
  float nrm = fmaxf(sqrtf(red[0]), 1e-12f);
  kn[(size_t)m * 128 + t] = (_Float16)(v / nrm);
}

// ---------------------------------------------------------------------------
// sim = qn (128x128) @ kn^T (128x400): WMMA, one wave per 16x16 tile,
// fragments gathered straight from global (rows are contiguous f16).
// ---------------------------------------------------------------------------
__global__ void sim_kernel(const _Float16* __restrict__ qn,
                           const _Float16* __restrict__ kn,
                           float* __restrict__ sim) {
  const int lane = threadIdx.x;                   // 32 threads = 1 wave
  const int nt = blockIdx.x;                      // 0..24 (key tiles)
  const int mt = blockIdx.y;                      // 0..7  (batch tiles)
  const int m    = lane & 15;
  const int half = lane >> 4;
  v8f acc = {};
#pragma unroll
  for (int kc = 0; kc < 4; ++kc) {
    const _Float16* qrow = qn + ((size_t)(mt * 16 + m) * 128 + kc * 32);
    v8h lo = *(const v8h*)(qrow + half * 16);
    v8h hi = *(const v8h*)(qrow + half * 16 + 8);
    v16h a;
#pragma unroll
    for (int i = 0; i < 8; ++i) { a[i] = lo[i]; a[i + 8] = hi[i]; }
    const _Float16* krow = kn + ((size_t)(nt * 16 + m) * 128 + kc * 32 + half * 16);
    v16h bf = *(const v16h*)krow;
    acc = __builtin_amdgcn_wmma_f32_16x16x32_f16(false, a, false, bf, (short)0, acc, false, false);
  }
#pragma unroll
  for (int r = 0; r < 8; ++r) {
    int row = mt * 16 + half * 8 + r;
    int col = nt * 16 + m;
    sim[(size_t)row * 400 + col] = acc[r];
  }
}

// ---------------------------------------------------------------------------
// Per-row top-5 (wave argmax, lower-index tie-break), softmax, gather values.
// ---------------------------------------------------------------------------
__global__ void topk_mem_kernel(const float* __restrict__ sim,
                                const float* __restrict__ values,
                                float* __restrict__ memv) {
  __shared__ float srow[400];
  __shared__ float tv[5];
  __shared__ int   ti[5];
  const int b = blockIdx.x;
  const int lane = threadIdx.x;                   // 32 threads
  for (int j = lane; j < 400; j += 32) srow[j] = sim[(size_t)b * 400 + j];
  __syncthreads();
  for (int k = 0; k < 5; ++k) {
    float bv = -3.4e38f; int bi = 0;
    for (int j = lane; j < 400; j += 32) {
      float v = srow[j];
      if (v > bv) { bv = v; bi = j; }
    }
#pragma unroll
    for (int off = 16; off > 0; off >>= 1) {
      float ov = __shfl_xor(bv, off);
      int   oi = __shfl_xor(bi, off);
      if (ov > bv || (ov == bv && oi < bi)) { bv = ov; bi = oi; }
    }
    if (lane == 0) { tv[k] = bv; ti[k] = bi; srow[bi] = -3.4e38f; }
    __syncthreads();
  }
  float w[5];
  float mx = tv[0];
#pragma unroll
  for (int k = 1; k < 5; ++k) mx = fmaxf(mx, tv[k]);
  float ssum = 0.0f;
#pragma unroll
  for (int k = 0; k < 5; ++k) { w[k] = __expf(tv[k] - mx); ssum += w[k]; }
  const float inv = 1.0f / ssum;
  for (int d = lane; d < 128; d += 32) {
    float s = 0.0f;
#pragma unroll
    for (int k = 0; k < 5; ++k) s += w[k] * inv * values[(size_t)ti[k] * 128 + d];
    memv[(size_t)b * 128 + d] = s;
  }
}

// ---------------------------------------------------------------------------
// Decoder: ConvT(128->64,k4,s2,p1) 1x1->2x2 + ReLU; ConvT(64->32) ->4x4 +
// ReLU; Conv(32->1,3x3,p1) ->4x4.  One block per batch, all in LDS.
// ---------------------------------------------------------------------------
__global__ void decode_kernel(const float* __restrict__ memv,
                              const float* __restrict__ wd1, const float* __restrict__ bd1,
                              const float* __restrict__ wd2, const float* __restrict__ bd2,
                              const float* __restrict__ wd3, const float* __restrict__ bd3,
                              float* __restrict__ out) {
  __shared__ float ms[128];
  __shared__ float h1s[64 * 4];                   // [oc][y*2+x]
  __shared__ float h2s[32 * 16];                  // [oc][y*4+x]
  const int b = blockIdx.x;
  const int t = threadIdx.x;                      // 128 threads
  ms[t] = memv[(size_t)b * 128 + t];
  __syncthreads();
  // stage 1: 1x1 -> 2x2 (only taps [y+1][x+1] contribute)
#pragma unroll
  for (int it = 0; it < 2; ++it) {
    int idx = t + it * 128;
    int oc = idx >> 2, p = idx & 3;
    int yy = p >> 1, xx = p & 1;
    float s = bd1[oc];
    for (int ic = 0; ic < 128; ++ic)
      s = fmaf(ms[ic], wd1[((ic * 64 + oc) * 4 + (yy + 1)) * 4 + (xx + 1)], s);
    h1s[oc * 4 + p] = fmaxf(s, 0.0f);
  }
  __syncthreads();
  // stage 2: 2x2 -> 4x4, ky = y+1-2*yi, kx = x+1-2*xi
#pragma unroll
  for (int it = 0; it < 4; ++it) {
    int idx = t + it * 128;
    int oc = idx >> 4, p = idx & 15;
    int y = p >> 2, x = p & 3;
    float s = bd2[oc];
    for (int ic = 0; ic < 64; ++ic) {
#pragma unroll
      for (int yi = 0; yi < 2; ++yi) {
        int ky = y + 1 - 2 * yi;
        if ((unsigned)ky >= 4u) continue;
#pragma unroll
        for (int xi = 0; xi < 2; ++xi) {
          int kx = x + 1 - 2 * xi;
          if ((unsigned)kx >= 4u) continue;
          s = fmaf(h1s[ic * 4 + yi * 2 + xi], wd2[((ic * 32 + oc) * 4 + ky) * 4 + kx], s);
        }
      }
    }
    h2s[oc * 16 + p] = fmaxf(s, 0.0f);
  }
  __syncthreads();
  // stage 3: 3x3 pad1 conv, 32 -> 1
  if (t < 16) {
    int y = t >> 2, x = t & 3;
    float s = bd3[0];
    for (int ic = 0; ic < 32; ++ic) {
#pragma unroll
      for (int dy = 0; dy < 3; ++dy) {
        int yy = y + dy - 1;
        if ((unsigned)yy >= 4u) continue;
#pragma unroll
        for (int dx = 0; dx < 3; ++dx) {
          int xx = x + dx - 1;
          if ((unsigned)xx >= 4u) continue;
          s = fmaf(h2s[ic * 16 + yy * 4 + xx], wd3[(ic * 3 + dy) * 3 + dx], s);
        }
      }
    }
    out[(size_t)b * 16 + t] = s;
  }
}

// ---------------------------------------------------------------------------
extern "C" void kernel_launch(void* const* d_in, const int* in_sizes, int n_in,
                              void* d_out, int out_size, void* d_ws, size_t ws_size,
                              hipStream_t stream) {
  (void)in_sizes; (void)n_in; (void)out_size; (void)ws_size;
  const float* x      = (const float*)d_in[0];
  const float* we1    = (const float*)d_in[1];
  const float* be1    = (const float*)d_in[2];
  const float* we2    = (const float*)d_in[3];
  const float* be2    = (const float*)d_in[4];
  const float* we3    = (const float*)d_in[5];
  const float* be3    = (const float*)d_in[6];
  const float* keys   = (const float*)d_in[7];
  const float* values = (const float*)d_in[8];
  const float* wd1    = (const float*)d_in[9];
  const float* bd1    = (const float*)d_in[10];
  const float* wd2    = (const float*)d_in[11];
  const float* bd2    = (const float*)d_in[12];
  const float* wd3    = (const float*)d_in[13];
  const float* bd3    = (const float*)d_in[14];
  float* out = (float*)d_out;

  const int B = 128;
  char* ws = (char*)d_ws;
  size_t off = 0;
  auto alloc = [&](size_t bytes) -> void* {
    void* p = ws + off;
    off = (off + bytes + 255) & ~(size_t)255;
    return p;
  };
  float2*   F    = (float2*)   alloc((size_t)B * 256 * 256 * sizeof(float2));        // 64 MB
  float*    xm   = (float*)    alloc((size_t)B * 256 * 256 * sizeof(float));         // 32 MB
  _Float16* h1   = (_Float16*) alloc((size_t)B * 128 * 128 * 32 * sizeof(_Float16)); // 134 MB
  _Float16* h2p  = (_Float16*) alloc((size_t)B * 64 * 64 * 64 * sizeof(_Float16));   // 67 MB
  _Float16* wt2  = (_Float16*) alloc((size_t)9 * 64 * 32 * sizeof(_Float16));
  _Float16* wt3  = (_Float16*) alloc((size_t)9 * 2 * 128 * 32 * sizeof(_Float16));
  float*    qsum = (float*)    alloc((size_t)B * 128 * sizeof(float));
  _Float16* qn   = (_Float16*) alloc((size_t)B * 128 * sizeof(_Float16));
  _Float16* kn   = (_Float16*) alloc((size_t)400 * 128 * sizeof(_Float16));
  float*    sim  = (float*)    alloc((size_t)B * 400 * sizeof(float));
  float*    memv = (float*)    alloc((size_t)B * 128 * sizeof(float));

  pack_w2_kernel<<<(9 * 64 * 32 + 255) / 256, 256, 0, stream>>>(we2, wt2);
  pack_w3_kernel<<<(9 * 2 * 128 * 32 + 255) / 256, 256, 0, stream>>>(we3, wt3);
  zero_kernel<<<(B * 128 + 255) / 256, 256, 0, stream>>>(qsum, B * 128);

  fft_rows_kernel<<<dim3(256, B), 128, 0, stream>>>(x, F);
  fft_cols_mag_kernel<<<dim3(256, B), 128, 0, stream>>>(F, xm);
  conv1_pool_kernel<<<dim3(8, 8, B), 256, 0, stream>>>(xm, we1, be1, h1);
  conv2_pool_kernel<<<dim3(16, 16, B), 256, 0, stream>>>(h1, wt2, be2, h2p);
  conv3_qsum_kernel<<<dim3(4, 64, B), 256, 0, stream>>>(h2p, wt3, be3, qsum);
  qnorm_kernel<<<B, 128, 0, stream>>>(qsum, qn);
  knorm_kernel<<<400, 128, 0, stream>>>(keys, kn);
  sim_kernel<<<dim3(25, 8), 32, 0, stream>>>(qn, kn, sim);
  topk_mem_kernel<<<B, 32, 0, stream>>>(sim, values, memv);
  decode_kernel<<<B, 128, 0, stream>>>(memv, wd1, bd1, wd2, bd2, wd3, bd3, out);
}
